// TreeAttentionWrapper_2216203124922
// MI455X (gfx1250) — matrix operand order repaired and checked by reading, not compiled, not run on hardware
//
#include <hip/hip_runtime.h>

// ---------------- problem constants ----------------
#define SEQ    2048
#define HEADS  16
#define DIM    64
#define QTILE  16                 // q columns per wave (N axis of S^T)
#define NWAVES 4
#define QBLK   (QTILE * NWAVES)   // 64 q rows per block
#define KTILE  64                 // keys per iteration
#define NSUB   4                  // 16-key subtiles per iteration
#define MROWB  72                 // padded LDS mask row stride (64B + 8B TDM pad)
#define NEG_INF (-3.0e38f)
#define M_INIT  (-1.0e30f)        // clamp so fully-masked tiles give p = exp2(-3e38) = 0
// softmax in exp2 domain: fold 1/sqrt(64) * log2(e) into Q
#define QSCALE (0.125f * 1.44269504088896340736f)

typedef __bf16 bf16_t;
typedef __attribute__((ext_vector_type(16))) __bf16        v16bf;
typedef __attribute__((ext_vector_type(8)))  float         v8f;
typedef __attribute__((ext_vector_type(4)))  float         f32x4;
typedef __attribute__((ext_vector_type(4)))  unsigned int  u32x4;
typedef __attribute__((ext_vector_type(2)))  unsigned int  u32x2;
typedef __attribute__((ext_vector_type(8)))  int           i32x8;
typedef __attribute__((ext_vector_type(4)))  int           i32x4;

union BF16x16 { v16bf v; u32x4 q[2]; };

#if defined(__has_builtin)
#  if __has_builtin(__builtin_amdgcn_tensor_load_to_lds)
#    define HAVE_TDM 1
#  endif
#endif
#ifndef HAVE_TDM
#  define HAVE_TDM 0
#endif

// ---- cross-half-wave exchange (lanes ^ 16) via ds_swizzle SWAPX16 ----
__device__ __forceinline__ unsigned swz16u(unsigned x) {
  return (unsigned)__builtin_amdgcn_ds_swizzle((int)x, 0x401F); // xor 0x10, and 0x1f
}
__device__ __forceinline__ float swz16f(float x) {
  return __builtin_bit_cast(
      float, __builtin_amdgcn_ds_swizzle(__builtin_bit_cast(int, x), 0x401F));
}
__device__ __forceinline__ unsigned pack2bf(float a, float b) {
  union { __bf16 h[2]; unsigned u; } x;
  x.h[0] = (bf16_t)a; x.h[1] = (bf16_t)b;
  return x.u;   // lowers to v_cvt_pk_bf16_f32
}

#if HAVE_TDM
// 2D TDM: 1-byte elems, tile KTILE x QBLK, row stride SEQ; LDS padding 2 dwords
// every 16 dwords -> 72B LDS row stride (bank-conflict-free b64 mask reads).
__device__ __forceinline__ void tdm_load_mask_tile(unsigned lds_off,
                                                   const unsigned char* gptr) {
  unsigned long long ga = (unsigned long long)gptr;
  u32x4 g0;
  g0[0] = 1u;                                            // count=1, user mode
  g0[1] = lds_off;
  g0[2] = (unsigned)(ga & 0xFFFFFFFFull);
  g0[3] = (unsigned)((ga >> 32) & 0x01FFFFFFull) | (2u << 30); // type=2
  const unsigned td0 = SEQ, td1 = SEQ, tl0 = KTILE, tl1 = QBLK;
  const unsigned long long st0 = SEQ;
  i32x8 g1;
  g1[0] = (int)((0u << 16)      // data_size = 1 byte
              | (1u << 20)      // pad_enable
              | (3u << 22)      // pad_interval: 16 dwords (64B)
              | (1u << 25));    // pad_amount: 2 dwords (8B)
  g1[1] = (int)((td0 & 0xFFFFu) << 16);
  g1[2] = (int)((td0 >> 16) | ((td1 & 0xFFFFu) << 16));
  g1[3] = (int)((td1 >> 16) | (tl0 << 16));
  g1[4] = (int)(tl1 & 0xFFFFu);
  g1[5] = (int)(st0 & 0xFFFFFFFFull);
  g1[6] = (int)((st0 >> 32) & 0xFFFFull);
  g1[7] = 0;
  i32x4 gz = {0, 0, 0, 0};
#if __clang_major__ >= 23
  i32x8 gz8 = {0, 0, 0, 0, 0, 0, 0, 0};
  __builtin_amdgcn_tensor_load_to_lds(g0, g1, gz, gz, gz8, 0);
#else
  __builtin_amdgcn_tensor_load_to_lds(g0, g1, gz, gz, 0);
#endif
}
#endif

// ---------------- staging: global -> regs, regs -> LDS (bf16) ----------------
__device__ __forceinline__ void stage_load(const float* kh, const float* vh,
                                           const unsigned char* mask,
                                           size_t mask_base, int kt, int t,
                                           f32x4* kf, f32x4* vf, u32x4* mv) {
  const int kkey = t >> 1, kd0 = (t & 1) * 32;        // K: 1 row half/thread
  const float* kp = kh + (size_t)(kt + kkey) * DIM + kd0;
#pragma unroll
  for (int i = 0; i < 8; ++i) kf[i] = *(const f32x4*)(kp + 4 * i);
  const int vkg = t >> 4, vdg = t & 15;               // V: 8 keys x 4 d /thread
  const float* vp = vh + (size_t)(kt + 8 * vkg) * DIM + 4 * vdg;
#pragma unroll
  for (int j = 0; j < 8; ++j) vf[j] = *(const f32x4*)(vp + (size_t)j * DIM);
  if (kt + 2 * KTILE < SEQ) {                         // warm HBM->L2
    __builtin_prefetch(kp + 2 * KTILE * DIM, 0, 1);
    __builtin_prefetch(vp + 2 * KTILE * DIM, 0, 1);
  }
#if !HAVE_TDM
  const int mr = t >> 1, mc = (t & 1) * 32;
  const unsigned char* mp = mask + mask_base + (size_t)mr * SEQ + kt + mc;
  mv[0] = *(const u32x4*)(mp);
  mv[1] = *(const u32x4*)(mp + 16);
#else
  (void)mask; (void)mask_base; (void)mv;
#endif
}

__device__ __forceinline__ void stage_commit(bf16_t* sKb, bf16_t* sVtb,
                                             unsigned char* sMb, int t,
                                             const f32x4* kf, const f32x4* vf,
                                             const u32x4* mv) {
  const int kkey = t >> 1, kd0 = (t & 1) * 32;
  {
    BF16x16 a, b;
#pragma unroll
    for (int i = 0; i < 16; ++i) {
      a.v[i] = (bf16_t)kf[i >> 2][i & 3];
      b.v[i] = (bf16_t)kf[4 + (i >> 2)][i & 3];
    }
    u32x4* dst = (u32x4*)(sKb + (size_t)kkey * DIM + kd0);
    dst[0] = a.q[0]; dst[1] = a.q[1]; dst[2] = b.q[0]; dst[3] = b.q[1];
  }
  const int vkg = t >> 4, vdg = t & 15;
#pragma unroll
  for (int i = 0; i < 4; ++i) {  // d row = 4*vdg+i, keys 8*vkg..+7 packed b128
    union { unsigned u[4]; u32x4 q; } pk;
#pragma unroll
    for (int j = 0; j < 4; ++j) pk.u[j] = pack2bf(vf[2 * j][i], vf[2 * j + 1][i]);
    *(u32x4*)(sVtb + (size_t)(4 * vdg + i) * KTILE + 8 * vkg) = pk.q;
  }
#if !HAVE_TDM
  const int mr = t >> 1, mc = (t & 1) * 32;   // replicate TDM's 72B padded rows
  unsigned long long* md = (unsigned long long*)(sMb + (size_t)mr * MROWB + mc);
  union { u32x4 q; unsigned long long d[2]; } a0, a1;
  a0.q = mv[0]; a1.q = mv[1];
  md[0] = a0.d[0]; md[1] = a0.d[1]; md[2] = a1.d[0]; md[3] = a1.d[1];
#else
  (void)sMb; (void)mv;
#endif
}

__global__ __launch_bounds__(128) void tree_attn_kernel(
    const float* __restrict__ q, const float* __restrict__ k,
    const float* __restrict__ v, const unsigned char* __restrict__ mask,
    float* __restrict__ out) {
  __shared__ __align__(16) bf16_t        sK[2][KTILE][DIM];        // 16 KB
  __shared__ __align__(16) bf16_t        sVt[2][DIM][KTILE];       // 16 KB
  __shared__ __align__(16) unsigned char sMraw[2][QBLK * MROWB];   // 9 KB

  const int t    = threadIdx.x;
  const int lane = t & 31;
  const int w    = t >> 5;
  const int n    = lane & 15;   // q column owned by this lane
  const int hi   = lane >> 4;

  const int qblk  = blockIdx.x;
  const int h     = blockIdx.y;
  const int qrow0 = qblk * QBLK;
  const size_t mask_base = (size_t)qrow0 * SEQ;

  const float* qh = q + (size_t)h * SEQ * DIM;
  const float* kh = k + (size_t)h * SEQ * DIM;
  const float* vh = v + (size_t)h * SEQ * DIM;
  float*       oh = out + (size_t)h * SEQ * DIM;

  // ---- Q^T B-fragments: lane=q column, halves = 16 consecutive d ----
  v16bf qb[2];
  {
    const float* qr = qh + (size_t)(qrow0 + w * QTILE + n) * DIM;
#pragma unroll
    for (int c = 0; c < 2; ++c) {
      const float* g0 = qr + 32 * c + 16 * hi;
      f32x4 f0 = *(const f32x4*)(g0);
      f32x4 f1 = *(const f32x4*)(g0 + 4);
      f32x4 f2 = *(const f32x4*)(g0 + 8);
      f32x4 f3 = *(const f32x4*)(g0 + 12);
      v16bf b;
#pragma unroll
      for (int i = 0; i < 4; ++i) {
        b[i]      = (bf16_t)(f0[i] * QSCALE);
        b[4 + i]  = (bf16_t)(f1[i] * QSCALE);
        b[8 + i]  = (bf16_t)(f2[i] * QSCALE);
        b[12 + i] = (bf16_t)(f3[i] * QSCALE);
      }
      qb[c] = b;
    }
  }

  BF16x16 ones;  // A operand for column-sum WMMA
#pragma unroll
  for (int i = 0; i < 16; ++i) ones.v[i] = (bf16_t)1.0f;

  float m = M_INIT, l = 0.f;   // per-lane (per-query) softmax state
  v8f oT[4];                   // O^T accumulators: slot=d row, lane=q col
#pragma unroll
  for (int dt = 0; dt < 4; ++dt)
#pragma unroll
    for (int r = 0; r < 8; ++r) oT[dt][r] = 0.f;

  // ---------------- software pipeline ----------------
  const int NIT = SEQ / KTILE;
  f32x4 kf[8], vf[8];
  u32x4 mv[2] = {{0, 0, 0, 0}, {0, 0, 0, 0}};
  stage_load(kh, vh, mask, mask_base, 0, t, kf, vf, mv);
#if HAVE_TDM
  if (t == 0)
    tdm_load_mask_tile((unsigned)(unsigned long long)&sMraw[0][0], mask + mask_base);
#endif
  stage_commit(&sK[0][0][0], &sVt[0][0][0], &sMraw[0][0], t, kf, vf, mv);
#if HAVE_TDM
  __builtin_amdgcn_s_wait_tensorcnt(0);
#endif
  __syncthreads();

  for (int it = 0; it < NIT; ++it) {
    const int kt  = it * KTILE;
    const int cur = it & 1, nxt = cur ^ 1;
    const bool have_next = (it + 1 < NIT);
    if (have_next) {
      stage_load(kh, vh, mask, mask_base, kt + KTILE, t, kf, vf, mv);
#if HAVE_TDM
      if (t == 0)
        tdm_load_mask_tile((unsigned)(unsigned long long)&sMraw[nxt][0],
                           mask + mask_base + kt + KTILE);
#endif
    }

    const bf16_t*        cK  = &sK[cur][0][0];
    const bf16_t*        cVt = &sVt[cur][0][0];
    const unsigned char* cM  = &sMraw[cur][0];

    // ---- S^T = K . Q^T : M=key, N=q (softmax dim lands in lanes) ----
    v8f st[NSUB];
#pragma unroll
    for (int s = 0; s < NSUB; ++s) {
      BF16x16 a0, a1;  // A frag of K subtile: row = key, halves = d runs
      const bf16_t* krow = cK + (size_t)(16 * s + n) * DIM;
      a0.q[0] = *(const u32x4*)(krow + 16 * hi);
      a0.q[1] = *(const u32x4*)(krow + 16 * hi + 8);
      a1.q[0] = *(const u32x4*)(krow + 32 + 16 * hi);
      a1.q[1] = *(const u32x4*)(krow + 32 + 16 * hi + 8);
      v8f z = {0.f, 0.f, 0.f, 0.f, 0.f, 0.f, 0.f, 0.f};
      v8f t0 = __builtin_amdgcn_wmma_f32_16x16x32_bf16(
          false, a0.v, false, qb[0], (short)0, z, false, false);
      st[s] = __builtin_amdgcn_wmma_f32_16x16x32_bf16(
          false, a1.v, false, qb[1], (short)0, t0, false, false);
    }

    // ---- mask (per-lane q row; padded 72B rows -> conflict-free b64) ----
    const unsigned char* mrow = cM + (size_t)(w * QTILE + n) * MROWB;
    u32x2 mw[NSUB];
#pragma unroll
    for (int s = 0; s < NSUB; ++s)
      mw[s] = *(const u32x2*)(mrow + 16 * s + 8 * hi);

    float smax = NEG_INF;
#pragma unroll
    for (int s = 0; s < NSUB; ++s)
#pragma unroll
      for (int r = 0; r < 8; ++r) {
        const unsigned by = (mw[s][r >> 2] >> (8 * (r & 3))) & 0xFFu;
        const float xs = by ? st[s][r] : NEG_INF;
        st[s][r] = xs;
        smax = fmaxf(smax, xs);
      }
    smax = fmaxf(smax, swz16f(smax));         // other 32 keys live in lane^16
    const float mnew = fmaxf(m, smax);        // m >= -1e30 => masked p -> 0
    const float corr = __builtin_amdgcn_exp2f(m - mnew);
    m = mnew;

#pragma unroll
    for (int dt = 0; dt < 4; ++dt)
#pragma unroll
      for (int r = 0; r < 8; ++r) oT[dt][r] *= corr;

    // ---- P^T B-frags in registers (pack + SWAPX16), l via ones-WMMA, PV ----
    float lsum = 0.f;
#pragma unroll
    for (int g = 0; g < 2; ++g) {
      unsigned pk_e[4], pk_o[4];
#pragma unroll
      for (int j = 0; j < 4; ++j) {
        const float e0 = __builtin_amdgcn_exp2f(st[2 * g][2 * j] - mnew);
        const float e1 = __builtin_amdgcn_exp2f(st[2 * g][2 * j + 1] - mnew);
        const float o0 = __builtin_amdgcn_exp2f(st[2 * g + 1][2 * j] - mnew);
        const float o1 = __builtin_amdgcn_exp2f(st[2 * g + 1][2 * j + 1] - mnew);
        pk_e[j] = pack2bf(e0, e1);
        pk_o[j] = pack2bf(o0, o1);
      }
      BF16x16 pB;
#pragma unroll
      for (int j = 0; j < 4; ++j) {
        const unsigned ex_e = swz16u(pk_e[j]);
        const unsigned ex_o = swz16u(pk_o[j]);
        pB.q[0][j] = hi ? ex_o    : pk_e[j];   // keys 32g+16hi+0..7
        pB.q[1][j] = hi ? pk_o[j] : ex_e;      // keys 32g+16hi+8..15
      }
      {  // l += column-sum(P^T) : ones(16x32) x P^T(32x16)
        v8f z = {0.f, 0.f, 0.f, 0.f, 0.f, 0.f, 0.f, 0.f};
        v8f rs = __builtin_amdgcn_wmma_f32_16x16x32_bf16(
            false, ones.v, false, pB.v, (short)0, z, false, false);
        lsum += rs[0];
      }
#pragma unroll
      for (int dt = 0; dt < 4; ++dt) {  // O^T += V^T . P^T
        BF16x16 va;
        const bf16_t* vrow = cVt + (size_t)(16 * dt + n) * KTILE + 32 * g;
        va.q[0] = *(const u32x4*)(vrow + 16 * hi);
        va.q[1] = *(const u32x4*)(vrow + 16 * hi + 8);
        oT[dt] = __builtin_amdgcn_wmma_f32_16x16x32_bf16(
            false, va.v, false, pB.v, (short)0, oT[dt], false, false);
      }
    }
    l = l * corr + lsum;

    if (have_next) {
      stage_commit(&sK[nxt][0][0], &sVt[nxt][0][0], &sMraw[nxt][0], t, kf, vf, mv);
#if HAVE_TDM
      __builtin_amdgcn_s_wait_tensorcnt(0);
#endif
    }
    __syncthreads();
  }

  // ---- epilogue: one q row per lane, coalesced b128 stores ----
  const float inv = 1.f / l;
  float* orow = oh + (size_t)(qrow0 + w * QTILE + n) * DIM;
#pragma unroll
  for (int dt = 0; dt < 4; ++dt) {
    f32x4 lo, hi4;
#pragma unroll
    for (int r = 0; r < 4; ++r) {
      lo[r]  = oT[dt][r] * inv;
      hi4[r] = oT[dt][4 + r] * inv;
    }
    *(f32x4*)(orow + 16 * dt + 8 * hi)     = lo;
    *(f32x4*)(orow + 16 * dt + 8 * hi + 4) = hi4;
  }
}

extern "C" void kernel_launch(void* const* d_in, const int* in_sizes, int n_in,
                              void* d_out, int out_size, void* d_ws,
                              size_t ws_size, hipStream_t stream) {
  const float* q = (const float*)d_in[0];
  const float* k = (const float*)d_in[1];
  const float* v = (const float*)d_in[2];
  const unsigned char* mask = (const unsigned char*)d_in[3];
  float* out = (float*)d_out;
  dim3 grid(SEQ / QBLK, HEADS);
  dim3 block(128);
  hipLaunchKernelGGL(tree_attn_kernel, grid, block, 0, stream, q, k, v, mask, out);
}
